// MoEExperts_7894149890291
// MI455X (gfx1250) — matrix operand order repaired and checked by reading, not compiled, not run on hardware
//
#include <hip/hip_runtime.h>
#include <hip/hip_bf16.h>
#include <math.h>

// ---------------- problem constants (from reference) ----------------
constexpr int NE   = 8;      // experts
constexpr int DD   = 2048;   // model dim
constexpr int HH   = 2048;   // expert hidden dim
constexpr int NT   = 4096;   // B*L tokens
constexpr int TOPK = 2;
constexpr int NP   = NT * TOPK;  // 8192 routed pairs
constexpr int CAP  = NP;         // worst-case pairs per expert

constexpr int BM  = 32;
constexpr int BN  = 64;
constexpr int BK  = 32;   // K step = WMMA K for bf16
constexpr int BKP = 40;   // LDS K stride (80 B: 16-byte aligned, bank-skewed)

typedef __attribute__((ext_vector_type(16))) __bf16 v16bf;
typedef __attribute__((ext_vector_type(8)))  __bf16 v8bf;
typedef __attribute__((ext_vector_type(8)))  float  v8f;

__device__ __forceinline__ unsigned pk_bf16(float a, float b) {
    union { __bf16 h[2]; unsigned u; } q;
    q.h[0] = (__bf16)a; q.h[1] = (__bf16)b;   // -> v_cvt_pk_bf16_f32
    return q.u;
}
__device__ __forceinline__ uint2 pk4(float4 v) {
    return make_uint2(pk_bf16(v.x, v.y), pk_bf16(v.z, v.w));
}
__device__ __forceinline__ v16bf cat16(v8bf lo, v8bf hi) {
    return __builtin_shufflevector(lo, hi, 0,1,2,3,4,5,6,7,8,9,10,11,12,13,14,15);
}
__device__ __forceinline__ float gelu_exact(float x) {
    return 0.5f * x * (1.0f + erff(x * 0.70710678118654752f));
}

// ---------------- kernel 0: zero out + counters ----------------
__global__ void moe_zero_kernel(float* out, int* counts, long long n_out) {
    long long i = (long long)blockIdx.x * blockDim.x + threadIdx.x;
    if (i < n_out) out[i] = 0.0f;
    if (i < NE) counts[i] = 0;
}

// ---------------- kernel 1: routing ----------------
__global__ void moe_route_kernel(const float* __restrict__ weights,
                                 const int*   __restrict__ expert_indices,
                                 const float* __restrict__ scale,
                                 int* __restrict__ counts,
                                 int* __restrict__ tok,
                                 float* __restrict__ cw) {
    int n = blockIdx.x * blockDim.x + threadIdx.x;
    if (n >= NT) return;
#pragma unroll
    for (int k = 0; k < TOPK; ++k) {
        int p = n * TOPK + k;
        int e = expert_indices[p];
        int pos = atomicAdd(&counts[e], 1);
        tok[e * CAP + pos] = p;
        cw[p] = weights[p] * scale[e];
    }
}

// ---------------- kernel 2: act = gelu(x @ Wg) * (x @ Wu) ----------------
// grid: (HH/BN, CAP/BM, NE), block 256
__global__ __launch_bounds__(256)
void moe_gateup_kernel(const float* __restrict__ x,
                       const float* __restrict__ gate_up,   // [E, D, 2H]
                       const int*   __restrict__ counts,
                       const int*   __restrict__ tok,
                       __bf16*      __restrict__ act) {     // [NP, H]
    const int e     = blockIdx.z;
    const int cnt   = counts[e];                   // uniform
    const int m0    = blockIdx.y * BM;
    if (m0 >= cnt) return;                         // uniform exit
    const int jbase = blockIdx.x * BN;

    const int* tokE = tok + (size_t)e * CAP;

    __shared__ __align__(16) __bf16 As[2][BM][BKP];   // row-major   [m][k]
    __shared__ __align__(16) __bf16 Bg[2][BN][BKP];   // transposed  [n][k]
    __shared__ __align__(16) __bf16 Bu[2][BN][BKP];
    __shared__ int spair[BM];                         // pair id (-1 invalid)

    const int tid   = threadIdx.x;
    const int lane  = tid & 31;
    const int wave  = tid >> 5;
    const int half  = lane >> 4;
    const int lrow  = lane & 15;
    const int mloc  = (wave & 1) * 16;
    const int nloc  = (wave >> 1) * 16;
    const int kbA   = half * 16;         // A frag: K in {kbA..kbA+15}
    const int kbB   = half * 16;         // B frag: K in {kbB..kbB+15}

    if (tid < BM) {                      // epilogue table, written once
        int gm = m0 + tid;
        spair[tid] = (gm < cnt) ? tokE[gm] : -1;
    }

    // ---- register-hoisted per-thread staging pointers ----
    const int ar  = tid >> 3;            // A: row 0..31
    const int ac4 = (tid & 7) * 4;       // A: 4-float column group
    const int bkp = tid >> 4;            // B: k-pair 0..15
    const int bc4 = (tid & 15) * 4;      // B: 4-col group

    int pA = (m0 + ar < cnt) ? tokE[m0 + ar] : -1;
    const float* aP = (pA >= 0) ? (x + (size_t)(pA >> 1) * DD + ac4) : nullptr;
    const float* gP = gate_up + ((size_t)e * DD + 2 * bkp) * (size_t)(2 * HH)
                      + jbase + bc4;

    // ---- prologue: stage tile 0 into buffer 0 ----
    {
        float4 vA = make_float4(0.f, 0.f, 0.f, 0.f);
        if (aP) vA = *(const float4*)aP;
        *(uint2*)&As[0][ar][ac4] = pk4(vA);
        float4 g0 = *(const float4*)(gP);
        float4 g1 = *(const float4*)(gP + 2 * HH);
        float4 u0 = *(const float4*)(gP + HH);
        float4 u1 = *(const float4*)(gP + HH + 2 * HH);
        const float *g0p = (const float*)&g0, *g1p = (const float*)&g1;
        const float *u0p = (const float*)&u0, *u1p = (const float*)&u1;
#pragma unroll
        for (int j = 0; j < 4; ++j) {
            *(unsigned*)&Bg[0][bc4 + j][2 * bkp] = pk_bf16(g0p[j], g1p[j]);
            *(unsigned*)&Bu[0][bc4 + j][2 * bkp] = pk_bf16(u0p[j], u1p[j]);
        }
    }
    __syncthreads();

    v8f cg = {};
    v8f cu = {};
    int buf = 0;

    for (int k0 = 0; k0 < DD; k0 += BK) {
        const int nxt = buf ^ 1;
        const bool hasNext = (k0 + BK < DD);

        // ---- issue next tile's global loads early (waits sink to use) ----
        float4 vA = make_float4(0.f, 0.f, 0.f, 0.f);
        float4 g0, g1, u0, u1;
        if (hasNext) {
            if (aP) vA = *(const float4*)(aP + k0 + BK);
            const float* gq = gP + (size_t)(k0 + BK) * (2 * HH);
            g0 = *(const float4*)(gq);
            g1 = *(const float4*)(gq + 2 * HH);
            u0 = *(const float4*)(gq + HH);
            u1 = *(const float4*)(gq + HH + 2 * HH);
            if (k0 + 2 * BK < DD)   // prefetch k+2 tile -> global_prefetch_b8
                __builtin_prefetch(gP + (size_t)(k0 + 2 * BK) * (2 * HH), 0, 3);
        }

        // ---- compute from current buffer ----
        v16bf a  = cat16(*(const v8bf*)&As[buf][mloc + lrow][kbA],
                         *(const v8bf*)&As[buf][mloc + lrow][kbA + 8]);
        v16bf fg = cat16(*(const v8bf*)&Bg[buf][nloc + lrow][kbB],
                         *(const v8bf*)&Bg[buf][nloc + lrow][kbB + 8]);
        v16bf fu = cat16(*(const v8bf*)&Bu[buf][nloc + lrow][kbB],
                         *(const v8bf*)&Bu[buf][nloc + lrow][kbB + 8]);
        cg = __builtin_amdgcn_wmma_f32_16x16x32_bf16(false, a, false, fg,
                                                     (short)0, cg, false, false);
        cu = __builtin_amdgcn_wmma_f32_16x16x32_bf16(false, a, false, fu,
                                                     (short)0, cu, false, false);

        // ---- convert + store next tile into alternate buffer ----
        if (hasNext) {
            *(uint2*)&As[nxt][ar][ac4] = pk4(vA);
            const float *g0p = (const float*)&g0, *g1p = (const float*)&g1;
            const float *u0p = (const float*)&u0, *u1p = (const float*)&u1;
#pragma unroll
            for (int j = 0; j < 4; ++j) {
                *(unsigned*)&Bg[nxt][bc4 + j][2 * bkp] = pk_bf16(g0p[j], g1p[j]);
                *(unsigned*)&Bu[nxt][bc4 + j][2 * bkp] = pk_bf16(u0p[j], u1p[j]);
            }
        }
        __syncthreads();
        buf = nxt;
    }

    // epilogue: act = gelu(gate) * up  (C layout: lane<16 -> M=r, else M=8+r)
#pragma unroll
    for (int r = 0; r < 8; ++r) {
        int rowin = r + half * 8;
        int p = spair[mloc + rowin];
        if (p >= 0) {
            float v = gelu_exact(cg[r]) * cu[r];
            act[(size_t)p * HH + jbase + nloc + lrow] = (__bf16)v;
        }
    }
}

// ---------------- kernel 3: out += cw * (act @ down) ----------------
// grid: (DD/BN, CAP/BM, NE), block 256
__global__ __launch_bounds__(256)
void moe_down_kernel(const __bf16* __restrict__ act,     // [NP, H]
                     const float*  __restrict__ down,    // [E, H, D]
                     const int*    __restrict__ counts,
                     const int*    __restrict__ tok,
                     const float*  __restrict__ cw,
                     float*        __restrict__ out) {   // [NT, D]
    const int e     = blockIdx.z;
    const int cnt   = counts[e];
    const int m0    = blockIdx.y * BM;
    if (m0 >= cnt) return;
    const int dbase = blockIdx.x * BN;

    const int* tokE = tok + (size_t)e * CAP;

    __shared__ __align__(16) __bf16 As[2][BM][BKP];
    __shared__ __align__(16) __bf16 Bs[2][BN][BKP];   // transposed [n][k]
    __shared__ int spair[BM];

    const int tid   = threadIdx.x;
    const int lane  = tid & 31;
    const int wave  = tid >> 5;
    const int half  = lane >> 4;
    const int lrow  = lane & 15;
    const int mloc  = (wave & 1) * 16;
    const int nloc  = (wave >> 1) * 16;
    const int kbA   = half * 16;
    const int kbB   = half * 16;

    if (tid < BM) {
        int gm = m0 + tid;
        spair[tid] = (gm < cnt) ? tokE[gm] : -1;
    }

    const int ar  = tid >> 3;
    const int ac4 = (tid & 7) * 4;
    const int bkp = tid >> 4;
    const int bc4 = (tid & 15) * 4;

    int pA = (m0 + ar < cnt) ? tokE[m0 + ar] : -1;
    const __bf16* aP = (pA >= 0) ? (act + (size_t)pA * HH + ac4) : nullptr;
    const float*  dP = down + ((size_t)e * HH + 2 * bkp) * (size_t)DD + dbase + bc4;

    // ---- prologue ----
    {
        uint2 vA = make_uint2(0u, 0u);
        if (aP) vA = *(const uint2*)aP;
        *(uint2*)&As[0][ar][ac4] = vA;
        float4 d0 = *(const float4*)(dP);
        float4 d1 = *(const float4*)(dP + DD);
        const float *d0p = (const float*)&d0, *d1p = (const float*)&d1;
#pragma unroll
        for (int j = 0; j < 4; ++j)
            *(unsigned*)&Bs[0][bc4 + j][2 * bkp] = pk_bf16(d0p[j], d1p[j]);
    }
    __syncthreads();

    v8f c = {};
    int buf = 0;

    for (int h0 = 0; h0 < HH; h0 += BK) {
        const int nxt = buf ^ 1;
        const bool hasNext = (h0 + BK < HH);

        uint2 vA = make_uint2(0u, 0u);
        float4 d0, d1;
        if (hasNext) {
            if (aP) vA = *(const uint2*)(aP + h0 + BK);
            const float* dq = dP + (size_t)(h0 + BK) * DD;
            d0 = *(const float4*)(dq);
            d1 = *(const float4*)(dq + DD);
            if (h0 + 2 * BK < HH)
                __builtin_prefetch(dP + (size_t)(h0 + 2 * BK) * DD, 0, 3);
        }

        v16bf a = cat16(*(const v8bf*)&As[buf][mloc + lrow][kbA],
                        *(const v8bf*)&As[buf][mloc + lrow][kbA + 8]);
        v16bf b = cat16(*(const v8bf*)&Bs[buf][nloc + lrow][kbB],
                        *(const v8bf*)&Bs[buf][nloc + lrow][kbB + 8]);
        c = __builtin_amdgcn_wmma_f32_16x16x32_bf16(false, a, false, b,
                                                    (short)0, c, false, false);

        if (hasNext) {
            *(uint2*)&As[nxt][ar][ac4] = vA;
            const float *d0p = (const float*)&d0, *d1p = (const float*)&d1;
#pragma unroll
            for (int j = 0; j < 4; ++j)
                *(unsigned*)&Bs[nxt][bc4 + j][2 * bkp] = pk_bf16(d0p[j], d1p[j]);
        }
        __syncthreads();
        buf = nxt;
    }

    // epilogue: weighted combine into out (tokens appear in <=2 expert lists)
#pragma unroll
    for (int r = 0; r < 8; ++r) {
        int rowin = r + half * 8;
        int p = spair[mloc + rowin];
        if (p >= 0) {
            int n = p >> 1;
            float w = cw[p];
            atomicAdd(&out[(size_t)n * DD + dbase + nloc + lrow], w * c[r]);
        }
    }
}

// ---------------- launch ----------------
extern "C" void kernel_launch(void* const* d_in, const int* in_sizes, int n_in,
                              void* d_out, int out_size, void* d_ws, size_t ws_size,
                              hipStream_t stream) {
    const float* x       = (const float*)d_in[0];   // [B,L,D]
    const float* weights = (const float*)d_in[1];   // [B,L,K]
    const int*   eidx    = (const int*)  d_in[2];   // [B,L,K]
    const float* gate_up = (const float*)d_in[3];   // [E,D,2H]
    const float* down    = (const float*)d_in[4];   // [E,H,D]
    const float* scale   = (const float*)d_in[5];   // [E]
    float* out = (float*)d_out;

    char* ws = (char*)d_ws;
    int*    counts = (int*)ws;
    int*    tok    = (int*)(ws + 256);
    float*  cw     = (float*)(ws + 256 + (size_t)NE * CAP * 4);
    __bf16* act    = (__bf16*)(ws + 256 + (size_t)NE * CAP * 4 + (size_t)NP * 4);

    const long long n_out = (long long)NT * DD;
    {
        int blocks = (int)((n_out + 255) / 256);
        moe_zero_kernel<<<blocks, 256, 0, stream>>>(out, counts, n_out);
    }
    moe_route_kernel<<<(NT + 255) / 256, 256, 0, stream>>>(weights, eidx, scale,
                                                           counts, tok, cw);
    {
        dim3 grid(HH / BN, CAP / BM, NE);
        moe_gateup_kernel<<<grid, 256, 0, stream>>>(x, gate_up, counts, tok, act);
    }
    {
        dim3 grid(DD / BN, CAP / BM, NE);
        moe_down_kernel<<<grid, 256, 0, stream>>>(act, down, counts, tok, cw, out);
    }
    (void)in_sizes; (void)n_in; (void)out_size; (void)ws_size;
}
